// TransformerAgent_48430051230161
// MI455X (gfx1250) — compile-verified, matrix operand order checked by reading, not run on hardware
//
#include <hip/hip_runtime.h>
#include <hip/hip_bf16.h>

// ---------------------------------------------------------------------------
// CDNA5 (gfx1250) bf16 WMMA implementation of the FAVOR-transformer agent.
// Wave32 everywhere. All matrix math via v_wmma_f32_16x16x32_bf16.
// ---------------------------------------------------------------------------

typedef __attribute__((ext_vector_type(4)))  unsigned int u32x4;
typedef __attribute__((ext_vector_type(16))) __bf16       v16bf;
typedef __attribute__((ext_vector_type(8)))  float        v8f;

union Frag16 { u32x4 q[2]; v16bf v; };
union Pack8  { u32x4 q;    unsigned short s[8];  };
union Pack32 { u32x4 q[4]; unsigned short s[32]; };

__device__ __forceinline__ float bf2f(unsigned short u) {
  union { unsigned int i; float f; } c; c.i = ((unsigned int)u) << 16; return c.f;
}
__device__ __forceinline__ unsigned short f2bf(float f) {
  union { float f; unsigned int i; } c; c.f = f;
  unsigned int r = c.i + 0x7FFFu + ((c.i >> 16) & 1u);   // round-to-nearest-even
  return (unsigned short)(r >> 16);
}

#define LDSP 40   // LDS row pitch in bf16 elems (80B rows -> 16B-aligned subrows)

// ---------------------------------------------------------------------------
// Generic GEMM: C(M x N) = epilogue(A(M x K) @ Wt(N x K)^T + bias)
// A, Wt, C bf16; bias fp32. Block = 128 threads (4 waves), tile 128x64:
// each wave owns 32 rows x 64 cols -> 8 WMMAs per K-chunk with B-frag reuse
// across two A-frags (1.5 ds_load_b128 per v_wmma).
// ep: 0 = bias, 1 = relu(bias)+1e-3 (FAVOR feature map), 2 = relu, 3 = 2*(..)
// ---------------------------------------------------------------------------
__global__ __launch_bounds__(128) void gemm_bf16(
    const unsigned short* __restrict__ A, const unsigned short* __restrict__ Wt,
    const float* __restrict__ bias, unsigned short* __restrict__ C,
    int N, int K, int ep)
{
  __shared__ unsigned short lA[128 * LDSP];
  __shared__ unsigned short lB[64 * LDSP];
  const int tid  = threadIdx.x;
  const int w    = tid >> 5, lane = tid & 31;
  const int rowBase = blockIdx.x * 128;
  const int colBase = blockIdx.y * 64;

  v8f acc[2][4] = {};

  for (int kc = 0; kc < K; kc += 32) {
#pragma unroll
    for (int p = 0; p < 4; ++p) {          // stage A tile: 128 rows x 32 cols
      int e = p * 1024 + tid * 8;
      int r = e >> 5, c = e & 31;
      *(u32x4*)&lA[r * LDSP + c] = *(const u32x4*)(A + (size_t)(rowBase + r) * K + kc + c);
    }
#pragma unroll
    for (int p = 0; p < 2; ++p) {          // stage B tile: 64 rows x 32 cols
      int e = p * 1024 + tid * 8;
      int r = e >> 5, c = e & 31;
      *(u32x4*)&lB[r * LDSP + c] = *(const u32x4*)(Wt + (size_t)(colBase + r) * K + kc + c);
    }
    if (kc + 32 < K) {  // gfx1250 global_prefetch_b8 for next K-chunk
      __builtin_prefetch(A  + (size_t)(rowBase + tid) * K + kc + 32, 0, 1);
      __builtin_prefetch(Wt + (size_t)(colBase + (tid >> 1)) * K + kc + 32, 0, 1);
    }
    __syncthreads();

    Frag16 fa[2];
#pragma unroll
    for (int mt = 0; mt < 2; ++mt) {
      const unsigned short* ap =
          &lA[(w * 32 + mt * 16 + (lane & 15)) * LDSP + ((lane >> 4) * 16)];
      fa[mt].q[0] = *(const u32x4*)ap;
      fa[mt].q[1] = *(const u32x4*)(ap + 8);
    }
#pragma unroll
    for (int nt = 0; nt < 4; ++nt) {
      Frag16 fb;
      const unsigned short* bp = &lB[(nt * 16 + (lane & 15)) * LDSP + ((lane >> 4) * 16)];
      fb.q[0] = *(const u32x4*)bp;
      fb.q[1] = *(const u32x4*)(bp + 8);
#pragma unroll
      for (int mt = 0; mt < 2; ++mt)
        acc[mt][nt] = __builtin_amdgcn_wmma_f32_16x16x32_bf16(
            false, fa[mt].v, false, fb.v, (short)0, acc[mt][nt], false, false);
    }
    __syncthreads();
  }

#pragma unroll
  for (int mt = 0; mt < 2; ++mt) {
#pragma unroll
    for (int nt = 0; nt < 4; ++nt) {
#pragma unroll
      for (int i = 0; i < 8; ++i) {
        int r = rowBase + w * 32 + mt * 16 + i + 8 * (lane >> 4);
        int c = colBase + nt * 16 + (lane & 15);
        float y = acc[mt][nt][i] + bias[c];
        if      (ep == 1) y = fmaxf(y, 0.f) + 1.0e-3f;
        else if (ep == 2) y = fmaxf(y, 0.f);
        else if (ep == 3) y = 2.f * y;
        C[(size_t)r * N + c] = f2bf(y);
      }
    }
  }
}

// ---------------------------------------------------------------------------
// attn_kv: per (b,h) compute kv[m][d] = sum_l kp[l][m]*v[l][d] (32x32) and
// ksum[m] = sum_l kp[l][m]. 4 waves split L=4096; WMMA over K-chunks of 32.
// kv stored transposed (d-major) in bf16 for use as B operand of attn_num.
// ---------------------------------------------------------------------------
__global__ __launch_bounds__(128) void attn_kv(
    const unsigned short* __restrict__ kp, const unsigned short* __restrict__ v,
    unsigned short* __restrict__ kvT, float* __restrict__ ksum)
{
  __shared__ unsigned short lKT[4][32 * LDSP];
  __shared__ unsigned short lVT[4][32 * LDSP];
  __shared__ float kvpart[4][32 * 32];
  __shared__ float kspart[4][32];
  const int tid = threadIdx.x, w = tid >> 5, lane = tid & 31;
  const int bh = blockIdx.x, b = bh >> 3, h = bh & 7;
  const size_t base = ((size_t)b * 4096) * 256 + (size_t)h * 32;

  v8f acc[2][2] = {};
  float ks = 0.f;

  for (int l0 = w * 1024; l0 < w * 1024 + 1024; l0 += 32) {
    Pack32 pk, pv;
    const unsigned short* kprow = kp + base + (size_t)(l0 + lane) * 256;
    const unsigned short* vrow  = v  + base + (size_t)(l0 + lane) * 256;
#pragma unroll
    for (int q = 0; q < 4; ++q) {
      pk.q[q] = *(const u32x4*)(kprow + q * 8);
      pv.q[q] = *(const u32x4*)(vrow  + q * 8);
    }
#pragma unroll
    for (int m = 0; m < 32; ++m) {          // transpose-store (lane = l index)
      lKT[w][m * LDSP + lane] = pk.s[m];
      lVT[w][m * LDSP + lane] = pv.s[m];
    }
    // per-wave LDS ops are in order: safe to read own wave's region
#pragma unroll
    for (int j = 0; j < 32; ++j) ks += bf2f(lKT[w][lane * LDSP + j]);

    Frag16 fa[2], fb[2];
#pragma unroll
    for (int t = 0; t < 2; ++t) {
      const unsigned short* ap = &lKT[w][(t * 16 + (lane & 15)) * LDSP + ((lane >> 4) * 16)];
      fa[t].q[0] = *(const u32x4*)ap; fa[t].q[1] = *(const u32x4*)(ap + 8);
      const unsigned short* bp = &lVT[w][(t * 16 + (lane & 15)) * LDSP + ((lane >> 4) * 16)];
      fb[t].q[0] = *(const u32x4*)bp; fb[t].q[1] = *(const u32x4*)(bp + 8);
    }
#pragma unroll
    for (int mt = 0; mt < 2; ++mt)
#pragma unroll
      for (int nt = 0; nt < 2; ++nt)
        acc[mt][nt] = __builtin_amdgcn_wmma_f32_16x16x32_bf16(
            false, fa[mt].v, false, fb[nt].v, (short)0, acc[mt][nt], false, false);
  }

#pragma unroll
  for (int mt = 0; mt < 2; ++mt)
#pragma unroll
    for (int nt = 0; nt < 2; ++nt)
#pragma unroll
      for (int i = 0; i < 8; ++i) {
        int m = mt * 16 + i + 8 * (lane >> 4);
        int d = nt * 16 + (lane & 15);
        kvpart[w][m * 32 + d] = acc[mt][nt][i];
      }
  kspart[w][lane] = ks;
  __syncthreads();

  if (w == 0) {
    int m = lane;
    ksum[bh * 32 + m] = kspart[0][m] + kspart[1][m] + kspart[2][m] + kspart[3][m];
    for (int d = 0; d < 32; ++d) {
      float s = kvpart[0][m * 32 + d] + kvpart[1][m * 32 + d] +
                kvpart[2][m * 32 + d] + kvpart[3][m * 32 + d];
      kvT[(size_t)bh * 1024 + d * 32 + m] = f2bf(s);   // [d][m] layout
    }
  }
}

// ---------------------------------------------------------------------------
// attn_num: per (row-tile, head): num = qp @ kv (K=32, one WMMA chunk),
// den = qp . ksum ; a = num/den written in place over qp.
// ---------------------------------------------------------------------------
__global__ __launch_bounds__(128) void attn_num(
    const unsigned short* __restrict__ kvT, const float* __restrict__ ksum,
    unsigned short* __restrict__ qp)
{
  __shared__ unsigned short lKV[32 * LDSP];
  __shared__ float lks[32];
  __shared__ float lden[64];
  const int tid = threadIdx.x, w = tid >> 5, lane = tid & 31;
  const int rowBase = blockIdx.x * 64;
  const int h = blockIdx.y;
  const int b = rowBase >> 12;
  const int bh = b * 8 + h;

  { // stage kv^T tile (32x32) into LDS, rows are d (= B-operand [n][k] layout)
    int e = tid * 8, d = e >> 5, m = e & 31;
    *(u32x4*)&lKV[d * LDSP + m] = *(const u32x4*)(kvT + (size_t)bh * 1024 + e);
  }
  if (tid < 32) lks[tid] = ksum[bh * 32 + tid];
  __syncthreads();

  if (tid < 64) {
    Pack32 pq;
    const unsigned short* qrow = qp + (size_t)(rowBase + tid) * 256 + h * 32;
#pragma unroll
    for (int q = 0; q < 4; ++q) pq.q[q] = *(const u32x4*)(qrow + q * 8);
    float den = 0.f;
#pragma unroll
    for (int m = 0; m < 32; ++m) den += bf2f(pq.s[m]) * lks[m];
    lden[tid] = den;
  }
  __syncthreads();

  Frag16 fa;
  const unsigned short* ap =
      qp + (size_t)(rowBase + w * 16 + (lane & 15)) * 256 + h * 32 + (lane >> 4) * 16;
  fa.q[0] = *(const u32x4*)ap;
  fa.q[1] = *(const u32x4*)(ap + 8);

  v8f acc[2] = {};
#pragma unroll
  for (int nt = 0; nt < 2; ++nt) {
    Frag16 fb;
    const unsigned short* bp = &lKV[(nt * 16 + (lane & 15)) * LDSP + ((lane >> 4) * 16)];
    fb.q[0] = *(const u32x4*)bp;
    fb.q[1] = *(const u32x4*)(bp + 8);
    acc[nt] = __builtin_amdgcn_wmma_f32_16x16x32_bf16(
        false, fa.v, false, fb.v, (short)0, acc[nt], false, false);
  }
#pragma unroll
  for (int nt = 0; nt < 2; ++nt)
#pragma unroll
    for (int i = 0; i < 8; ++i) {
      int r = w * 16 + i + 8 * (lane >> 4);
      int c = nt * 16 + (lane & 15);
      float y = acc[nt][i] / lden[r];
      qp[(size_t)(rowBase + r) * 256 + h * 32 + c] = f2bf(y);
    }
}

// ---------------------------------------------------------------------------
// LayerNorm over 256-elem rows. One wave per row, wave32 shuffle reductions.
// ---------------------------------------------------------------------------
__global__ __launch_bounds__(256) void ln_rows(
    const unsigned short* __restrict__ src, const float* __restrict__ gamma,
    const float* __restrict__ beta, unsigned short* __restrict__ dst)
{
  const int tid = threadIdx.x, w = tid >> 5, lane = tid & 31;
  const size_t row = (size_t)blockIdx.x * 8 + w;
  Pack8 p; p.q = *(const u32x4*)(src + row * 256 + lane * 8);
  float f[8]; float s = 0.f;
#pragma unroll
  for (int j = 0; j < 8; ++j) { f[j] = bf2f(p.s[j]); s += f[j]; }
#pragma unroll
  for (int o = 16; o > 0; o >>= 1) s += __shfl_xor(s, o, 32);
  float mean = s * (1.f / 256.f);
  float vs = 0.f;
#pragma unroll
  for (int j = 0; j < 8; ++j) { float d = f[j] - mean; vs += d * d; }
#pragma unroll
  for (int o = 16; o > 0; o >>= 1) vs += __shfl_xor(vs, o, 32);
  float inv = rsqrtf(vs * (1.f / 256.f) + 1e-6f);
  Pack8 outp;
#pragma unroll
  for (int j = 0; j < 8; ++j) {
    int c = lane * 8 + j;
    outp.s[j] = f2bf((f[j] - mean) * inv * gamma[c] + beta[c]);
  }
  *(u32x4*)(dst + row * 256 + lane * 8) = outp.q;
}

// ---------------------------------------------------------------------------
// Weight fp32 -> bf16 transposed copy: dst(N x K) = src(K x N)^T
// ---------------------------------------------------------------------------
__global__ void wconv_t(const float* __restrict__ src, unsigned short* __restrict__ dst,
                        int K, int N)
{
  int idx = blockIdx.x * 256 + threadIdx.x;
  if (idx >= K * N) return;
  int n = idx / K, k = idx - n * K;
  dst[idx] = f2bf(src[(size_t)k * N + n]);
}

// Pad ins into (B, 4096, 128) bf16 with row l==0 zeroed.
__global__ void prep_xin(const float* __restrict__ ins, unsigned short* __restrict__ Xin)
{
  int idx = blockIdx.x * 256 + threadIdx.x;
  if (idx >= 16 * 4096 * 128) return;
  int c = idx & 127;
  int bl = idx >> 7;
  int l = bl & 4095, b = bl >> 12;
  float val = 0.f;
  if (l > 0) val = ins[((size_t)b * 4095 + (l - 1)) * 128 + c];
  Xin[idx] = f2bf(val);
}

// Overwrite x[b, 0, :] = resets[b] ? 0 : hidden_state[b, 0, :]
__global__ void fix_row0(const float* __restrict__ hidden,
                         const unsigned char* __restrict__ resets,
                         unsigned short* __restrict__ x)
{
  int b = blockIdx.x, j = threadIdx.x;
  float hv = resets[b] ? 0.f : hidden[b * 256 + j];
  x[(size_t)b * 4096 * 256 + j] = f2bf(hv);
}

// new_hidden = x[:,0,:] (fp32); q_vals = new_hidden @ qp_W + qp_b
__global__ __launch_bounds__(256) void final_out(
    const unsigned short* __restrict__ x, const float* __restrict__ qpW,
    const float* __restrict__ qpb, float* __restrict__ out)
{
  __shared__ float hbuf[256];
  int b = blockIdx.x, j = threadIdx.x;
  float hv = bf2f(x[(size_t)b * 4096 * 256 + j]);
  out[b * 256 + j] = hv;
  hbuf[j] = hv;
  __syncthreads();
  if (j < 32) {
    float acc = qpb[j];
    for (int k = 0; k < 256; ++k) acc += hbuf[k] * qpW[k * 32 + j];
    out[4096 + b * 32 + j] = acc;
  }
}

// ---------------------------------------------------------------------------
// Host launch
// ---------------------------------------------------------------------------
static const size_t kWSub[6] = {0, 131072, 262144, 393216, 524288, 1048576}; // bytes
// 0=WqT 1=WkT 2=WvT 3=WoT (65536 bf16 each), 4=W1T 5=W2T (262144 bf16 each)

extern "C" void kernel_launch(void* const* d_in, const int* in_sizes, int n_in,
                              void* d_out, int out_size, void* d_ws, size_t ws_size,
                              hipStream_t stream)
{
  (void)in_sizes; (void)n_in; (void)out_size; (void)ws_size;
  const float* hidden = (const float*)d_in[0];
  const float* ins    = (const float*)d_in[1];
  const unsigned char* resets = (const unsigned char*)d_in[2];
  const float* emb_W  = (const float*)d_in[3];
  const float* emb_b  = (const float*)d_in[4];
  const float* Wq = (const float*)d_in[5];
  const float* bq = (const float*)d_in[6];
  const float* Wk = (const float*)d_in[7];
  const float* bk = (const float*)d_in[8];
  const float* Wv = (const float*)d_in[9];
  const float* bv = (const float*)d_in[10];
  const float* Wo = (const float*)d_in[11];
  const float* bo = (const float*)d_in[12];
  const float* ln1s = (const float*)d_in[13];
  const float* ln1b = (const float*)d_in[14];
  const float* W1 = (const float*)d_in[15];
  const float* b1 = (const float*)d_in[16];
  const float* W2 = (const float*)d_in[17];
  const float* b2 = (const float*)d_in[18];
  const float* ln2s = (const float*)d_in[19];
  const float* ln2b = (const float*)d_in[20];
  const float* qpW = (const float*)d_in[21];
  const float* qpb = (const float*)d_in[22];
  float* out = (float*)d_out;

  char* ws = (char*)d_ws;
  const size_t MB = (size_t)1 << 20;
  unsigned short* embT = (unsigned short*)(ws + 0);            // 64KB
  unsigned short* Xin  = (unsigned short*)(ws + 8 * MB);       // 16MB
  unsigned short* X    = (unsigned short*)(ws + 24 * MB);      // 32MB
  unsigned short* QP   = (unsigned short*)(ws + 56 * MB);      // 32MB (later holds 'a')
  unsigned short* KP   = (unsigned short*)(ws + 88 * MB);      // 32MB
  unsigned short* V    = (unsigned short*)(ws + 120 * MB);     // 32MB
  unsigned short* T0   = (unsigned short*)(ws + 152 * MB);     // 32MB
  unsigned short* H    = (unsigned short*)(ws + 184 * MB);     // 128MB
  unsigned short* KVT  = (unsigned short*)(ws + 312 * MB);     // 256KB
  float*          KS   = (float*)(ws + 312 * MB + 262144);     // 16KB

  auto WT = [&](int layer, int which) -> unsigned short* {
    return (unsigned short*)(ws + 65536 + (size_t)layer * 1572864 + kWSub[which]);
  };
  auto launch_t = [&](const float* src, unsigned short* dst, int K, int N) {
    int total = K * N;
    wconv_t<<<dim3((total + 255) / 256), dim3(256), 0, stream>>>(src, dst, K, N);
  };

  // weight conversion / transposition (bf16, N x K layout)
  launch_t(emb_W, embT, 128, 256);
  for (int L = 0; L < 4; ++L) {
    launch_t(Wq + (size_t)L * 65536,  WT(L, 0), 256, 256);
    launch_t(Wk + (size_t)L * 65536,  WT(L, 1), 256, 256);
    launch_t(Wv + (size_t)L * 65536,  WT(L, 2), 256, 256);
    launch_t(Wo + (size_t)L * 65536,  WT(L, 3), 256, 256);
    launch_t(W1 + (size_t)L * 262144, WT(L, 4), 256, 1024);
    launch_t(W2 + (size_t)L * 262144, WT(L, 5), 1024, 256);
  }

  // input embedding: x = pad(ins) @ emb_W + emb_b ; row0 = hidden/reset
  prep_xin<<<dim3(32768), dim3(256), 0, stream>>>(ins, Xin);
  gemm_bf16<<<dim3(512, 4), dim3(128), 0, stream>>>(Xin, embT, emb_b, X, 256, 128, 0);
  fix_row0<<<dim3(16), dim3(256), 0, stream>>>(hidden, resets, X);

  for (int L = 0; L < 4; ++L) {
    // qp = relu(x@Wq + bq)+eps ; kp likewise ; v = x@Wv + bv
    gemm_bf16<<<dim3(512, 4), dim3(128), 0, stream>>>(X, WT(L, 0), bq + L * 256, QP, 256, 256, 1);
    gemm_bf16<<<dim3(512, 4), dim3(128), 0, stream>>>(X, WT(L, 1), bk + L * 256, KP, 256, 256, 1);
    gemm_bf16<<<dim3(512, 4), dim3(128), 0, stream>>>(X, WT(L, 2), bv + L * 256, V,  256, 256, 0);
    // kv / ksum reductions, then a = (qp@kv)/(qp.ksum) in place over QP
    attn_kv<<<dim3(128), dim3(128), 0, stream>>>(KP, V, KVT, KS);
    attn_num<<<dim3(1024, 8), dim3(128), 0, stream>>>(KVT, KS, QP);
    // o-proj, x = 2*o, LN1
    gemm_bf16<<<dim3(512, 4), dim3(128), 0, stream>>>(QP, WT(L, 3), bo + L * 256, T0, 256, 256, 3);
    ln_rows<<<dim3(8192), dim3(256), 0, stream>>>(T0, ln1s + L * 256, ln1b + L * 256, X);
    // FFN: h = relu(x@W1+b1); t0 = 2*(h@W2+b2); LN2 -> x
    gemm_bf16<<<dim3(512, 16), dim3(128), 0, stream>>>(X, WT(L, 4), b1 + L * 1024, H, 1024, 256, 2);
    gemm_bf16<<<dim3(512, 4),  dim3(128), 0, stream>>>(H, WT(L, 5), b2 + L * 256,  T0, 256, 1024, 3);
    ln_rows<<<dim3(8192), dim3(256), 0, stream>>>(T0, ln2s + L * 256, ln2b + L * 256, X);
  }

  final_out<<<dim3(16), dim3(256), 0, stream>>>(X, qpW, qpb, out);
}
